// STN_RNN_11527692222524
// MI455X (gfx1250) — matrix operand order repaired and checked by reading, not compiled, not run on hardware
//
#include <hip/hip_runtime.h>
#include <hip/hip_bf16.h>

// STN-RNN (leaky LSTM) persistent-kernel for gfx1250.
// B=32, T=8192, IN=1, H=128. grid=2 blocks x 256 threads (8 waves, wave32).
// Each block owns 16 batch rows; each wave owns 16 hidden columns across all
// 4 gates. W_hh held in VGPRs as bf16 WMMA B-tiles; h state double-buffered
// in LDS; gates computed with v_wmma_f32_16x16x32_bf16.

typedef __attribute__((ext_vector_type(16))) __bf16 v16bf;
typedef __attribute__((ext_vector_type(8)))  float  v8f;

#define B_SZ 32
#define T_SZ 8192
#define H_SZ 128
#define ROWB 136   // padded LDS row length in bf16 elements (272 B) -> conflict-free b128 loads

__device__ __forceinline__ float fexp_(float x) {
  return __builtin_amdgcn_exp2f(x * 1.44269504088896340736f);
}
__device__ __forceinline__ float sig_(float x) {
  return __builtin_amdgcn_rcpf(1.0f + fexp_(-x));
}
__device__ __forceinline__ float tanh_(float x) {
  return 1.0f - 2.0f * __builtin_amdgcn_rcpf(fexp_(2.0f * x) + 1.0f);
}

__global__ __launch_bounds__(256)
void stn_lstm_kernel(const float* __restrict__ x,
                     const float* __restrict__ W_ih,
                     const float* __restrict__ W_hh,
                     const float* __restrict__ b_ih,
                     const float* __restrict__ b_hh,
                     float* __restrict__ out)
{
  __shared__ __align__(16) __bf16 hbuf[2][16][ROWB];

  const int tid   = threadIdx.x;
  const int wave  = tid >> 5;
  const int lane  = tid & 31;
  const int col   = lane & 15;     // tile column / A row (0..15)
  const int half  = lane >> 4;     // 0/1: which K/M half this lane covers
  const int j     = wave * 16 + col;          // hidden column 0..127
  const int b_base = blockIdx.x * 16;         // 16 batches per block
  const int rowoff = half * 8;                // C/D tile row offset for this lane

  // ---- zero both h double-buffers (state s0 = 0) ----
  {
    unsigned* p = (unsigned*)&hbuf[0][0][0];
    const int n = (int)(sizeof(hbuf) / 4);
    for (int i = tid; i < n; i += 256) p[i] = 0u;
  }

  // ---- preload W_hh^T as 16 bf16 B-tiles (4 gates x 4 k-steps) into VGPRs ----
  // B tile (32x16, K x N): lane holds column n = j of gate g; lanes 0-15 K=k0..k0+15,
  // lanes 16-31 K=k0+16..k0+31; element e of v16bf = K = kbase + e (2 packed / dword).
  v16bf Bt[16];
  #pragma unroll
  for (int g = 0; g < 4; ++g) {
    #pragma unroll
    for (int kk = 0; kk < 4; ++kk) {
      const int n     = g * 128 + j;
      const int kbase = kk * 32 + half * 16;
      const float* src = W_hh + n * H_SZ + kbase;
      v16bf bt;
      #pragma unroll
      for (int e = 0; e < 16; ++e) bt[e] = (__bf16)src[e];
      Bt[g * 4 + kk] = bt;
    }
  }

  // ---- per-lane bias and W_ih column constants (IN == 1) ----
  float biasv[4], wihv[4];
  #pragma unroll
  for (int g = 0; g < 4; ++g) {
    const int n = g * 128 + j;
    biasv[g] = b_ih[n] + b_hh[n];
    wihv[g]  = W_ih[n];
  }

  // ---- leaky state (f32) kept in registers; layout matches WMMA C/D tile ----
  float hs[8], cs[8];
  #pragma unroll
  for (int r = 0; r < 8; ++r) { hs[r] = 0.f; cs[r] = 0.f; }
  const float kleak = 0.5f;   // 1 / OS_FACTOR

  // ---- streaming pointers ----
  const float* xbase[8];
  float*       op[8];
  #pragma unroll
  for (int r = 0; r < 8; ++r) {
    const int b = b_base + rowoff + r;
    xbase[r] = x + (size_t)b * T_SZ;
    op[r]    = out + (size_t)b * T_SZ * H_SZ + j;
  }

  #pragma unroll 1
  for (int t = 0; t < T_SZ; ++t) {
    __syncthreads();   // prev step's h stores visible; prev reads done (double buffer)

    // ---- load A tiles (h state, 16x32 bf16 per k-step) from LDS ----
    const __bf16* hb = &hbuf[t & 1][0][0];
    union { v16bf v; uint4 q[2]; } A[4];
    #pragma unroll
    for (int kk = 0; kk < 4; ++kk) {
      const int k0 = kk * 32 + half * 16;
      const __bf16* pa = hb + col * ROWB + k0;
      A[kk].q[0] = *(const uint4*)(pa);
      A[kk].q[1] = *(const uint4*)(pa + 8);
    }

    // ---- x_t (broadcast within half-wave; L2-resident stream) ----
    float xr[8];
    #pragma unroll
    for (int r = 0; r < 8; ++r) xr[r] = xbase[r][t];
    __builtin_prefetch(&xbase[t & 7][(t + 256 < T_SZ) ? t + 256 : t], 0, 1);

    // ---- gates = h @ W_hh^T + (bias + x*W_ih) via chained WMMA ----
    v8f acc[4];
    #pragma unroll
    for (int g = 0; g < 4; ++g) {
      v8f c;
      #pragma unroll
      for (int r = 0; r < 8; ++r) c[r] = biasv[g] + xr[r] * wihv[g];
      c = __builtin_amdgcn_wmma_f32_16x16x32_bf16(false, A[0].v, false, Bt[g*4+0], (short)0, c, false, false);
      c = __builtin_amdgcn_wmma_f32_16x16x32_bf16(false, A[1].v, false, Bt[g*4+1], (short)0, c, false, false);
      c = __builtin_amdgcn_wmma_f32_16x16x32_bf16(false, A[2].v, false, Bt[g*4+2], (short)0, c, false, false);
      c = __builtin_amdgcn_wmma_f32_16x16x32_bf16(false, A[3].v, false, Bt[g*4+3], (short)0, c, false, false);
      acc[g] = c;
    }

    // ---- LSTM cell + leaky state update; emit h to LDS (next step) + output ----
    __bf16* hn_buf = &hbuf[(t + 1) & 1][0][0];
    #pragma unroll
    for (int r = 0; r < 8; ++r) {
      const float iv = sig_(acc[0][r]);
      const float fv = sig_(acc[1][r]);
      const float gv = tanh_(acc[2][r]);
      const float ov = sig_(acc[3][r]);
      const float cn = fv * cs[r] + iv * gv;
      const float hn = ov * tanh_(cn);
      const float hsn = hs[r] + kleak * (hn - hs[r]);
      const float csn = cs[r] + kleak * (cn - cs[r]);
      hs[r] = hsn; cs[r] = csn;
      hn_buf[(rowoff + r) * ROWB + j] = (__bf16)hsn;
      __builtin_nontemporal_store(hsn, op[r]);
      op[r] += H_SZ;
    }
  }

  // ---- final state: s_final[b][0:128]=h, [128:256]=c ----
  float* sf = out + (size_t)B_SZ * T_SZ * H_SZ;
  #pragma unroll
  for (int r = 0; r < 8; ++r) {
    const int b = b_base + rowoff + r;
    sf[(size_t)b * 2 * H_SZ + j]        = hs[r];
    sf[(size_t)b * 2 * H_SZ + H_SZ + j] = cs[r];
  }
}

extern "C" void kernel_launch(void* const* d_in, const int* in_sizes, int n_in,
                              void* d_out, int out_size, void* d_ws, size_t ws_size,
                              hipStream_t stream) {
  const float* x    = (const float*)d_in[0];
  const float* W_ih = (const float*)d_in[1];
  const float* W_hh = (const float*)d_in[2];
  const float* b_ih = (const float*)d_in[3];
  const float* b_hh = (const float*)d_in[4];
  float* out = (float*)d_out;
  stn_lstm_kernel<<<dim3(2), dim3(256), 0, stream>>>(x, W_ih, W_hh, b_ih, b_hh, out);
}
